// STU_78116865179927
// MI455X (gfx1250) — compile-verified, matrix operand order checked by reading, not run on hardware
//
#include <hip/hip_runtime.h>
#include <hip/hip_bf16.h>

// ---------------------------------------------------------------------------
// STU spectral conv for MI455X (gfx1250, wave32, WMMA, async LDS copy)
//   out = irfft( rfft(x@Mi) * G ),  G[f] = (V[f] + conj(V[N/2-f]))/N
// Kernel 1: split-bf16 WMMA GEMM  x_proj = x @ M_inputs   (fp32-accurate, 3 WMMAs)
// Kernel 2: filter spectra G (768 radix-4 FFTs of 16384, LDS-resident)
// Kernel 3: per-(b, channel-pair) radix-4 FFT conv; global<->LDS via async ops
// ---------------------------------------------------------------------------

typedef __attribute__((ext_vector_type(16))) __bf16        v16bf;
typedef __attribute__((ext_vector_type(8)))  float         v8f;
typedef __attribute__((ext_vector_type(8)))  unsigned short usht8;

#define B_DIM   4
#define L_DIM   8192
#define D_DIM   768
#define NFFT    16384
#define NHALF   8192
#define GSTRIDE 8224          // 8193 bins padded for alignment
#define TWN     4096          // twiddle table covers e in [0, N/4)

// ---------------- helpers ----------------
__device__ __forceinline__ unsigned short f2bf_rne(float f) {
    unsigned u = __float_as_uint(f);
    unsigned r = u + 0x7FFFu + ((u >> 16) & 1u);
    return (unsigned short)(r >> 16);
}
__device__ __forceinline__ float bf2f(unsigned short h) {
    return __uint_as_float(((unsigned)h) << 16);
}

union Frag { v16bf v; usht8 u[2]; };

// ---------------- Kernel 1: GEMM (M=32768, N=768, K=768) -------------------
// block tile 128x64, 8 waves (4x2), wave tile 32x32 = 2x2 WMMA 16x16 tiles.
#define BM 128
#define BN 64
#define BK 32
#define AST 40   // LDS row stride (ushorts), padded vs 32 to spread banks
#define BST 40

__global__ __launch_bounds__(256)
void stu_gemm_bf16x3(const float* __restrict__ X, const float* __restrict__ W,
                     float* __restrict__ Y) {
    __shared__ __align__(16) unsigned short Ah[BM * AST];
    __shared__ __align__(16) unsigned short Al[BM * AST];
    __shared__ __align__(16) unsigned short Bh[BN * BST];
    __shared__ __align__(16) unsigned short Bl[BN * BST];

    const int tid  = threadIdx.x;
    const int lane = tid & 31;
    const int wid  = tid >> 5;
    const int wm   = wid & 3;      // 0..3  (M direction)
    const int wn   = wid >> 2;     // 0..1  (N direction)
    const int h    = lane >> 4;    // half-wave selects K group
    const int ml   = lane & 15;    // row (A) / col (B) within tile
    const int m0   = blockIdx.x * BM;
    const int n0   = blockIdx.y * BN;

    v8f acc[2][2];
    const v8f vzero = {0.f,0.f,0.f,0.f,0.f,0.f,0.f,0.f};
    acc[0][0] = vzero; acc[0][1] = vzero; acc[1][0] = vzero; acc[1][1] = vzero;

    for (int k0 = 0; k0 < D_DIM; k0 += BK) {
        // stage A tile 128x32 fp32 -> bf16 hi/lo (packed b32 LDS stores)
        #pragma unroll
        for (int i = 0; i < 4; ++i) {
            int e = tid + i * 256;              // float4 id 0..1023
            int r = e >> 3, q = e & 7;
            const float4 v = *(const float4*)(X + (size_t)(m0 + r) * D_DIM + k0 + q * 4);
            float xs[4] = {v.x, v.y, v.z, v.w};
            unsigned short hi[4], lo[4];
            #pragma unroll
            for (int c = 0; c < 4; ++c) {
                hi[c] = f2bf_rne(xs[c]);
                lo[c] = f2bf_rne(xs[c] - bf2f(hi[c]));
            }
            int base = r * AST + q * 4;         // even -> 4B aligned
            *(unsigned*)&Ah[base]     = (unsigned)hi[0] | ((unsigned)hi[1] << 16);
            *(unsigned*)&Ah[base + 2] = (unsigned)hi[2] | ((unsigned)hi[3] << 16);
            *(unsigned*)&Al[base]     = (unsigned)lo[0] | ((unsigned)lo[1] << 16);
            *(unsigned*)&Al[base + 2] = (unsigned)lo[2] | ((unsigned)lo[3] << 16);
        }
        // stage B tile 32x64 fp32 -> bf16 hi/lo, transposed (col-major rows)
        #pragma unroll
        for (int i = 0; i < 2; ++i) {
            int e = tid + i * 256;              // float4 id 0..511
            int kr = e >> 4, q = e & 15;
            const float4 v = *(const float4*)(W + (size_t)(k0 + kr) * D_DIM + n0 + q * 4);
            float xs[4] = {v.x, v.y, v.z, v.w};
            #pragma unroll
            for (int c = 0; c < 4; ++c) {
                int n = q * 4 + c;
                unsigned short hi = f2bf_rne(xs[c]);
                Bh[n * BST + kr] = hi;
                Bl[n * BST + kr] = f2bf_rne(xs[c] - bf2f(hi));
            }
        }
        __syncthreads();

        Frag aH[2], aL[2], bH[2], bL[2];
        #pragma unroll
        for (int tm = 0; tm < 2; ++tm) {
            int m = wm * 32 + tm * 16 + ml;
            int base = m * AST + h * 8;
            aH[tm].u[0] = *(const usht8*)&Ah[base];
            aH[tm].u[1] = *(const usht8*)&Ah[base + 16];
            aL[tm].u[0] = *(const usht8*)&Al[base];
            aL[tm].u[1] = *(const usht8*)&Al[base + 16];
        }
        #pragma unroll
        for (int tn = 0; tn < 2; ++tn) {
            int n = wn * 32 + tn * 16 + ml;
            int base = n * BST + h * 8;
            bH[tn].u[0] = *(const usht8*)&Bh[base];
            bH[tn].u[1] = *(const usht8*)&Bh[base + 16];
            bL[tn].u[0] = *(const usht8*)&Bl[base];
            bL[tn].u[1] = *(const usht8*)&Bl[base + 16];
        }
        #pragma unroll
        for (int tm = 0; tm < 2; ++tm) {
            #pragma unroll
            for (int tn = 0; tn < 2; ++tn) {
                acc[tm][tn] = __builtin_amdgcn_wmma_f32_16x16x32_bf16(
                    false, aH[tm].v, false, bH[tn].v, (short)0, acc[tm][tn], false, false);
                acc[tm][tn] = __builtin_amdgcn_wmma_f32_16x16x32_bf16(
                    false, aH[tm].v, false, bL[tn].v, (short)0, acc[tm][tn], false, false);
                acc[tm][tn] = __builtin_amdgcn_wmma_f32_16x16x32_bf16(
                    false, aL[tm].v, false, bH[tn].v, (short)0, acc[tm][tn], false, false);
            }
        }
        __syncthreads();
    }

    #pragma unroll
    for (int tm = 0; tm < 2; ++tm) {
        #pragma unroll
        for (int tn = 0; tn < 2; ++tn) {
            int m = m0 + wm * 32 + tm * 16 + h * 8;
            int n = n0 + wn * 32 + tn * 16 + ml;
            #pragma unroll
            for (int r = 0; r < 8; ++r)
                Y[(size_t)(m + r) * D_DIM + n] = acc[tm][tn][r];
        }
    }
}

// ---------------- complex helpers -----------------------------------------
__device__ __forceinline__ float2 cadd(float2 a, float2 b) { return make_float2(a.x + b.x, a.y + b.y); }
__device__ __forceinline__ float2 csub(float2 a, float2 b) { return make_float2(a.x - b.x, a.y - b.y); }
__device__ __forceinline__ float2 cmul(float2 a, float2 b) {
    return make_float2(a.x * b.x - a.y * b.y, a.x * b.y + a.y * b.x);
}
// W_N^e from quarter-period table, e in [0, 3N/4)
__device__ __forceinline__ float2 twd(const float2* T, int e) {
    float2 t = T[e & (TWN - 1)];
    int qq = e >> 12;
    if (qq == 1)      t = make_float2(t.y, -t.x);   // * exp(-i*pi/2)
    else if (qq == 2) t = make_float2(-t.x, -t.y);  // * exp(-i*pi)
    return t;
}
__device__ __forceinline__ float2 twdc(const float2* T, int e) {   // conj(W_N^e)
    float2 t = twd(T, e);
    return make_float2(t.x, -t.y);
}
// base-4 digit reversal of 7 digits (14 bits)
__device__ __forceinline__ int dr4(int x) {
    unsigned t = __brev((unsigned)x) >> 18;
    return (int)(((t & 0x1555u) << 1) | ((t >> 1) & 0x1555u));
}
// build quarter-period twiddle table W_N^e = exp(-2*pi*i*e/N), e < N/4
__device__ __forceinline__ void build_tw(float2* T, int tid) {
    const float w0 = -6.28318530717958647692f / (float)NFFT;
    for (int e = tid; e < TWN; e += 256) {
        float sn, cs;
        sincosf(w0 * (float)e, &sn, &cs);
        T[e] = make_float2(cs, sn);
    }
}
// low 32 bits of a generic pointer to __shared__ = LDS byte offset
__device__ __forceinline__ unsigned lds_off32(const void* p) {
    return (unsigned)(size_t)p;
}

// ---------------- radix-4 FFT machinery (in LDS) ---------------------------
// Forward DIF-4: natural in -> base-4 digit-reversed out. 7 stages.
__device__ void fft4_fwd(float2* z, const float2* T, int tid) {
    for (int s = 0; s < 7; ++s) {
        const int qsh = 2 * (6 - s);       // log2(quarter)
        const int q   = 1 << qsh;
        for (int t = tid; t < 4096; t += 256) {
            int j    = t & (q - 1);
            int blk  = t >> qsh;
            int base = (blk << (qsh + 2)) + j;
            float2 a0 = z[base], a1 = z[base + q], a2 = z[base + 2*q], a3 = z[base + 3*q];
            float2 t0 = cadd(a0, a2), t1 = csub(a0, a2);
            float2 t2 = cadd(a1, a3), t3 = csub(a1, a3);
            float2 B0 = cadd(t0, t2);
            float2 B1 = make_float2(t1.x + t3.y, t1.y - t3.x);   // t1 - i*t3
            float2 B2 = csub(t0, t2);
            float2 B3 = make_float2(t1.x - t3.y, t1.y + t3.x);   // t1 + i*t3
            int e1 = j << (2 * s);                               // j * (N/m)
            z[base]       = B0;
            z[base +   q] = cmul(B1, twd(T, e1));
            z[base + 2*q] = cmul(B2, twd(T, 2 * e1));
            z[base + 3*q] = cmul(B3, twd(T, 3 * e1));
        }
        __syncthreads();
    }
}
// Inverse DIT-4: digit-reversed in -> natural out (un-normalized, factor N).
__device__ void fft4_inv(float2* z, const float2* T, int tid) {
    for (int ss = 0; ss < 7; ++ss) {
        const int qsh = 2 * ss;
        const int q   = 1 << qsh;
        for (int t = tid; t < 4096; t += 256) {
            int j    = t & (q - 1);
            int blk  = t >> qsh;
            int base = (blk << (qsh + 2)) + j;
            int e1   = j << (2 * (6 - ss));
            float2 B0 = z[base];
            float2 B1 = cmul(z[base +   q], twdc(T, e1));
            float2 B2 = cmul(z[base + 2*q], twdc(T, 2 * e1));
            float2 B3 = cmul(z[base + 3*q], twdc(T, 3 * e1));
            float2 u0 = cadd(B0, B2), u1 = csub(B0, B2);
            float2 u2 = cadd(B1, B3), u3 = csub(B1, B3);
            z[base]       = cadd(u0, u2);
            z[base +   q] = make_float2(u1.x - u3.y, u1.y + u3.x);  // u1 + i*u3
            z[base + 2*q] = csub(u0, u2);
            z[base + 3*q] = make_float2(u1.x + u3.y, u1.y - u3.x);  // u1 - i*u3
        }
        __syncthreads();
    }
}

// ---------------- Kernel 2: filter spectra G --------------------------------
// One WG per channel d: v = phi @ Mf[:,d]; V = FFT(v, 16384);
// G = (V[f] + conj(V[8192-f])) / NFFT   (irfft norm folded in here)
__global__ __launch_bounds__(256)
void stu_filter_fft(const float* __restrict__ phi, const float* __restrict__ Mf,
                    float2* __restrict__ G) {
    __shared__ float2 z[NFFT];
    __shared__ float2 T[TWN];
    const int d = blockIdx.x, tid = threadIdx.x;

    float mfr[24];                        // d uniform -> scalar loads
    #pragma unroll
    for (int k = 0; k < 24; ++k) mfr[k] = Mf[k * D_DIM + d];

    build_tw(T, tid);
    for (int l = tid; l < L_DIM; l += 256) {
        const float* pr = phi + (size_t)l * 24;
        float acc = 0.f;
        #pragma unroll
        for (int k = 0; k < 24; ++k) acc += pr[k] * mfr[k];
        z[l]         = make_float2(acc, 0.f);
        z[l + NHALF] = make_float2(0.f, 0.f);
    }
    __syncthreads();
    fft4_fwd(z, T, tid);
    const float nrm = 1.0f / (float)NFFT;
    float2* gd = G + (size_t)d * GSTRIDE;
    for (int f = tid; f <= NHALF; f += 256) {
        float2 V  = z[dr4(f)];
        float2 Vc = z[dr4(NHALF - f)];
        gd[f] = make_float2((V.x + Vc.x) * nrm, (V.y - Vc.y) * nrm);
    }
}

// ---------------- Kernel 3: conv for a (batch, channel-pair) ----------------
__global__ __launch_bounds__(256)
void stu_conv_fft(const float* __restrict__ U, const float2* __restrict__ G,
                  float* __restrict__ out) {
    __shared__ float2 z[NFFT];
    __shared__ float2 T[TWN];
    const int g = blockIdx.x;               // 0..1535
    const int b = g / (D_DIM / 2);
    const int da = (g % (D_DIM / 2)) * 2;   // even channel: pack (da, da+1)
    const int tid = threadIdx.x;

    // 1) kick off async global->LDS copies of (u_a, u_b) pairs (ASYNCcnt)
    const float* ub = U + (size_t)b * L_DIM * D_DIM + da;
    for (int l = tid; l < L_DIM; l += 256) {
        const float2* gp = (const float2*)(ub + (size_t)l * D_DIM);
        unsigned zoff = lds_off32(&z[l]);
        asm volatile("global_load_async_to_lds_b64 %0, %1, off"
                     :: "v"(zoff), "v"(gp) : "memory");
    }
    // 2) overlap: prefetch the G spectra this WG will consume (~128 KB)
    const float2* Ga = G + (size_t)da * GSTRIDE;
    const float2* Gb = Ga + GSTRIDE;
    for (int f = tid * 16; f <= NHALF; f += 256 * 16) {
        __builtin_prefetch(Ga + f, 0, 2);
        __builtin_prefetch(Gb + f, 0, 2);
    }
    // 3) overlap: twiddle table + zero-pad upper half (disjoint LDS region)
    build_tw(T, tid);
    for (int l = tid; l < NHALF; l += 256) z[l + NHALF] = make_float2(0.f, 0.f);
    // 4) wait for async copies, then barrier so all lanes see LDS
    asm volatile("s_wait_asynccnt 0x0" ::: "memory");
    __syncthreads();

    fft4_fwd(z, T, tid);

    // pointwise in digit-reversed order; each thread owns conjugate pair (f, N-f)
    for (int f = tid; f <= NHALF; f += 256) {
        int fr = (NFFT - f) & (NFFT - 1);
        int p0 = dr4(f), p1 = dr4(fr);
        float2 Z0 = z[p0], Z1 = z[p1];
        // unpack two real spectra: A = (Z0+conj(Z1))/2 ; Bq = (Z0-conj(Z1))*(-i/2)
        float Ax = 0.5f * (Z0.x + Z1.x), Ay = 0.5f * (Z0.y - Z1.y);
        float Bx = 0.5f * (Z0.y + Z1.y), By = 0.5f * (Z1.x - Z0.x);
        float2 ga = Ga[f], gb = Gb[f];
        float Yax = Ax * ga.x - Ay * ga.y, Yay = Ax * ga.y + Ay * ga.x;
        float Ybx = Bx * gb.x - By * gb.y, Yby = Bx * gb.y + By * gb.x;
        // repack: W[f] = Ya + i*Yb ; W[N-f] = conj(Ya) + i*conj(Yb)
        z[p0] = make_float2(Yax - Yby, Yay + Ybx);
        if (p1 != p0) z[p1] = make_float2(Yax + Yby, Ybx - Yay);
    }
    __syncthreads();
    fft4_inv(z, T, tid);
    // fft4_inv ends with a barrier: all LDS writes visible; 1/N already in G.

    // 5) async LDS->global stores of the result pairs; S_ENDPGM waits idle.
    float* ob = out + (size_t)b * L_DIM * D_DIM + da;
    for (int l = tid; l < L_DIM; l += 256) {
        float2* gp = (float2*)(ob + (size_t)l * D_DIM);
        unsigned zoff = lds_off32(&z[l]);
        asm volatile("global_store_async_from_lds_b64 %0, %1, off"
                     :: "v"(gp), "v"(zoff) : "memory");
    }
}

// ---------------- launch ----------------------------------------------------
extern "C" void kernel_launch(void* const* d_in, const int* in_sizes, int n_in,
                              void* d_out, int out_size, void* d_ws, size_t ws_size,
                              hipStream_t stream) {
    const float* x   = (const float*)d_in[0];
    const float* phi = (const float*)d_in[1];
    const float* Mi  = (const float*)d_in[2];
    const float* Mf  = (const float*)d_in[3];
    float* out = (float*)d_out;

    float*  xproj = (float*)d_ws;                                   // 96 MB
    float2* G = (float2*)((char*)d_ws + (size_t)B_DIM * L_DIM * D_DIM * sizeof(float));

    dim3 g1((B_DIM * L_DIM) / BM, D_DIM / BN);   // 256 x 12
    stu_gemm_bf16x3<<<g1, 256, 0, stream>>>(x, Mi, xproj);
    stu_filter_fft<<<D_DIM, 256, 0, stream>>>(phi, Mf, G);
    stu_conv_fft<<<B_DIM * (D_DIM / 2), 256, 0, stream>>>(xproj, G, out);
}
